// MultiHeadAttentionLayer_59313498357888
// MI455X (gfx1250) — matrix operand order repaired and checked by reading, not compiled, not run on hardware
//
#include <hip/hip_runtime.h>
#include <hip/hip_bf16.h>

#define DEV_INLINE __device__ __forceinline__

typedef __attribute__((ext_vector_type(16))) __bf16 v16bf;
typedef __attribute__((ext_vector_type(8)))  float  v8f;

constexpr int N_NODES = 10000;
constexpr int IN_DIM  = 256;
constexpr int NHEAD   = 8;
constexpr int DHEAD   = 32;
constexpr int HD      = NHEAD * DHEAD;  // 256
constexpr int OUT3    = 3 * HD;         // 768 (Q|K|V fused columns)
constexpr int MTILES  = N_NODES / 16;   // 625
constexpr int CTILES  = OUT3 / 16;      // 48
constexpr int TILES   = MTILES * CTILES;

// float -> bf16, round to nearest even
DEV_INLINE unsigned short f2bf(float f) {
  unsigned int u = __float_as_uint(f);
  u += 0x7FFFu + ((u >> 16) & 1u);
  return (unsigned short)(u >> 16);
}

// ---------------- prep kernels ----------------

__global__ void zero_accum_kernel(float* __restrict__ out, float* __restrict__ z) {
  int i = blockIdx.x * blockDim.x + threadIdx.x;
  if (i < N_NODES * HD)    out[i] = 0.0f;
  if (i < N_NODES * NHEAD) z[i]   = 0.0f;
}

__global__ void cvt_h_kernel(const float* __restrict__ h,
                             unsigned short* __restrict__ hbf) {
  int i = blockIdx.x * blockDim.x + threadIdx.x;
  if (i < N_NODES * IN_DIM) hbf[i] = f2bf(h[i]);
}

// Build WT[768][256] bf16 = transposed, fused [Wq|Wk|Wv]; bfull[768] = [bq|bk|bv]
__global__ void prep_w_kernel(const float* __restrict__ Wq, const float* __restrict__ Wk,
                              const float* __restrict__ Wv, const float* __restrict__ bq,
                              const float* __restrict__ bk, const float* __restrict__ bv,
                              unsigned short* __restrict__ WT, float* __restrict__ bfull) {
  int i = blockIdx.x * blockDim.x + threadIdx.x;
  if (i >= OUT3 * IN_DIM) return;
  int c = i / IN_DIM;           // fused output channel 0..767
  int k = i % IN_DIM;           // input feature
  int which = c / HD;           // 0=Q, 1=K, 2=V
  int cc = c % HD;
  const float* W = (which == 0) ? Wq : (which == 1) ? Wk : Wv;
  WT[i] = f2bf(W[(size_t)k * HD + cc]);
  if (k == 0) {
    const float* b = (which == 0) ? bq : (which == 1) ? bk : bv;
    bfull[c] = b[cc];
  }
}

// ---------------- fused QKV GEMM via v_wmma_f32_16x16x32_bf16 ----------------
// QKV[n][0:256)=Q, [256:512)=K, [512:768)=V  (f32, bias added)

__global__ void __launch_bounds__(256) qkv_gemm_kernel(
    const unsigned short* __restrict__ hbf,   // [N][256] bf16 row-major
    const unsigned short* __restrict__ WT,    // [768][256] bf16 row-major (pre-transposed)
    const float* __restrict__ bfull,          // [768]
    float* __restrict__ QKV) {                // [N][768] f32
  const int lane = threadIdx.x & 31;
  const int wv   = threadIdx.x >> 5;
  const int t    = blockIdx.x * 8 + wv;       // one 16x16 tile per wave
  if (t >= TILES) return;
  const int mtile = t % MTILES;
  const int ctile = t / MTILES;
  const int half  = lane >> 4;                // ISA layout: lane group 0/1
  const int l16   = lane & 15;
  const int arow  = mtile * 16 + l16;         // A row M = lane%16
  const int ncol  = ctile * 16 + l16;         // B/C/D column N = lane%16

  const uint4* aptr = (const uint4*)(hbf + (size_t)arow * IN_DIM);  // 8 bf16 / uint4
  const uint4* bptr = (const uint4*)(WT  + (size_t)ncol * IN_DIM);

  v8f acc = {};
#pragma unroll
  for (int kb = 0; kb < IN_DIM; kb += 32) {
    union { uint4 u[2]; v16bf v; } A, B;
    // A 16x32 bf16: halfword e in [0,8): K = kb + 8*half + e   (contiguous run)
    //               halfword e in [8,16): K = kb + 16 + 8*half + (e-8)
    A.u[0] = aptr[(kb >> 3) + half];
    A.u[1] = aptr[(kb >> 3) + 2 + half];
    // B 32x16 bf16: same K order as A (e<8: kb + 8*half + e; e>=8: kb + 16 + 8*half + (e-8))
    const int bi = (kb >> 3) + half;
    B.u[0] = bptr[bi];
    B.u[1] = bptr[bi + 2];
    acc = __builtin_amdgcn_wmma_f32_16x16x32_bf16(
        /*neg_a=*/false, A.v, /*neg_b=*/false, B.v,
        /*c_mod=*/(short)0, acc, /*reuse_a=*/false, /*reuse_b=*/false);
  }

  // C/D layout: VGPR r holds row (r + 8*half), column = lane%16
  const float bias = bfull[ncol];
  float* orow = QKV + (size_t)(mtile * 16 + 8 * half) * OUT3 + ncol;
#pragma unroll
  for (int r = 0; r < 8; ++r)
    orow[(size_t)r * OUT3] = acc[r] + bias;
}

// ---------------- edge phase: one wave32 per edge, lane = head dim ----------------

__global__ void __launch_bounds__(256) edge_kernel(
    const float* __restrict__ QKV, const int* __restrict__ src,
    const int* __restrict__ dst, float* __restrict__ wV,
    float* __restrict__ z, int E) {
  const int lane = threadIdx.x & 31;
  const int e = (int)((blockIdx.x * (unsigned)blockDim.x + threadIdx.x) >> 5);
  if (e >= E) return;
  const int s = src[e];
  const int d = dst[e];
  const float* Qrow = QKV + (size_t)d * OUT3;            // Q[dst]
  const float* Krow = QKV + (size_t)s * OUT3 + HD;       // K[src]
  const float* Vrow = QKV + (size_t)s * OUT3 + 2 * HD;   // V[src]
  const float rsD = 0.17677669529663687f;                // 1/sqrt(32)
#pragma unroll
  for (int h = 0; h < NHEAD; ++h) {
    const int c = h * DHEAD + lane;
    float partial = Krow[c] * Qrow[c];
#pragma unroll
    for (int off = 16; off > 0; off >>= 1)
      partial += __shfl_xor(partial, off, 32);
    float sc = partial * rsD;
    sc = fminf(5.0f, fmaxf(-5.0f, sc));
    const float p = __expf(sc);
    atomicAdd(&wV[(size_t)d * HD + c], p * Vrow[c]);     // coalesced f32 atomics (L2)
    if (lane == 0) atomicAdd(&z[d * NHEAD + h], p);
  }
}

// ---------------- normalize: wV / z, 0/0 -> 0 ----------------

__global__ void norm_kernel(float* __restrict__ out, const float* __restrict__ z) {
  int i = blockIdx.x * blockDim.x + threadIdx.x;
  if (i >= N_NODES * HD) return;
  int n = i / HD;
  int h = (i % HD) / DHEAD;
  float zz = z[n * NHEAD + h];
  out[i] = (zz > 0.0f) ? out[i] / zz : 0.0f;   // exp(..)>0, so z==0 iff no in-edges
}

extern "C" void kernel_launch(void* const* d_in, const int* in_sizes, int n_in,
                              void* d_out, int out_size, void* d_ws, size_t ws_size,
                              hipStream_t stream) {
  const float* h  = (const float*)d_in[0];
  const float* Wq = (const float*)d_in[1];
  const float* bq = (const float*)d_in[2];
  const float* Wk = (const float*)d_in[3];
  const float* bk = (const float*)d_in[4];
  const float* Wv = (const float*)d_in[5];
  const float* bv = (const float*)d_in[6];
  const int*   src = (const int*)d_in[7];
  const int*   dst = (const int*)d_in[8];
  const int E = in_sizes[7];
  float* out = (float*)d_out;

  // workspace carve-out (all offsets 16B-aligned): ~36.6 MB total
  char* ws = (char*)d_ws;
  size_t off = 0;
  unsigned short* hbf = (unsigned short*)(ws + off); off += (size_t)N_NODES * IN_DIM * 2;
  unsigned short* WT  = (unsigned short*)(ws + off); off += (size_t)OUT3 * IN_DIM * 2;
  float* bfull = (float*)(ws + off); off += (size_t)OUT3 * 4;
  float* QKV   = (float*)(ws + off); off += (size_t)N_NODES * OUT3 * 4;
  float* z     = (float*)(ws + off); off += (size_t)N_NODES * NHEAD * 4;

  zero_accum_kernel<<<(N_NODES * HD + 255) / 256, 256, 0, stream>>>(out, z);
  cvt_h_kernel<<<(N_NODES * IN_DIM + 255) / 256, 256, 0, stream>>>(h, hbf);
  prep_w_kernel<<<(OUT3 * IN_DIM + 255) / 256, 256, 0, stream>>>(
      Wq, Wk, Wv, bq, bk, bv, WT, bfull);
  qkv_gemm_kernel<<<(TILES + 7) / 8, 256, 0, stream>>>(hbf, WT, bfull, QKV);
  edge_kernel<<<(E * 32 + 255) / 256, 256, 0, stream>>>(QKV, src, dst, out, z, E);
  norm_kernel<<<(N_NODES * HD + 255) / 256, 256, 0, stream>>>(out, z);
}
